// CNN_LSTM_GCN_20873541058993
// MI455X (gfx1250) — matrix operand order repaired and checked by reading, not compiled, not run on hardware
//
#include <hip/hip_runtime.h>
#include <math.h>

// ---------------- problem constants (fixed by the reference) ----------------
#define N_NODES   10000
#define SEQ_LEN   32
#define NUM_FEAT  128
#define GCN_H     32
#define CONV_H    64
#define LSTM_H    128
#define TS        31                      // SEQ_LEN - 1
#define M_GCN     (N_NODES * SEQ_LEN)     // 320000 rows (multiple of 16)
#define M_CONV    (N_NODES * TS)          // 310000 rows (multiple of 16)

typedef __attribute__((ext_vector_type(16))) _Float16 v16h;
typedef __attribute__((ext_vector_type(8)))  float    v8f;

// ---------------- WMMA helpers (CDNA5 wave32, 16x16x32 f16 -> f32) ----------
__device__ __forceinline__ v8f wmma_f16(v16h a, v16h b, v8f c) {
  return __builtin_amdgcn_wmma_f32_16x16x32_f16(
      /*neg_a=*/false, a, /*neg_b=*/false, b,
      /*c_mod=*/(short)0, c, /*reuse_a=*/false, /*reuse_b=*/false);
}

// A fragment (16x32 f16): lane L holds row M=L&15; halves 0..7 = K kb..kb+7,
// halves 8..15 = K 16+kb..16+kb+7 where kb=(L>>4)*8.  (ISA 7.12.2)
__device__ __forceinline__ v16h a_frag_f16(const _Float16* rowp, int k0, int lane) {
  const int kb = ((lane >> 4) & 1) * 8;
  v16h a;
#pragma unroll
  for (int j = 0; j < 8; ++j) {
    a[j]     = rowp[k0 + kb + j];
    a[j + 8] = rowp[k0 + 16 + kb + j];
  }
  return a;
}

__device__ __forceinline__ v16h a_frag_f32(const float* rowp, int k0, int lane) {
  const int kb = ((lane >> 4) & 1) * 8;
  v16h a;
#pragma unroll
  for (int j = 0; j < 8; ++j) {
    a[j]     = (_Float16)rowp[k0 + kb + j];
    a[j + 8] = (_Float16)rowp[k0 + 16 + kb + j];
  }
  return a;
}

// B fragment from a PRE-PACKED weight buffer: fragments stored in register
// order, so each lane reads 16 contiguous halves (32 bytes -> 2x b128).
// Tile order: tile = kt * ntiles_n + nt ; within tile: [lane][16 halves].
__device__ __forceinline__ v16h b_frag_packed(const _Float16* __restrict__ Bp,
                                              int ntiles_n, int kt, int nt, int lane) {
  const size_t tile = (size_t)kt * ntiles_n + nt;
  return *(const v16h*)(Bp + (tile * 32 + (size_t)lane) * 16);
}

// ---------------- small prep kernels ----------------------------------------
__global__ void k_init_deg(int* deg) {             // self-loop => start at 1
  int i = blockIdx.x * blockDim.x + threadIdx.x;
  if (i < N_NODES) deg[i] = 1;
}
__global__ void k_deg_count(const int* __restrict__ dst, int E, int* __restrict__ deg) {
  int e = blockIdx.x * blockDim.x + threadIdx.x;
  if (e < E) atomicAdd(&deg[dst[e]], 1);
}
__global__ void k_dinv(const int* __restrict__ deg, float* __restrict__ dinv) {
  int i = blockIdx.x * blockDim.x + threadIdx.x;
  if (i < N_NODES) dinv[i] = rsqrtf((float)deg[i]);
}

// f32 [K][N] -> f16 [K][N]
__global__ void k_cvt(const float* __restrict__ w, _Float16* __restrict__ o, int n) {
  int i = blockIdx.x * blockDim.x + threadIdx.x;
  if (i < n) o[i] = (_Float16)w[i];
}
// f32 [rows][cols] -> f16 [cols][rows]   (transpose w_ih / w_hh)
__global__ void k_cvt_T(const float* __restrict__ w, _Float16* __restrict__ o,
                        int rows, int cols) {
  int i = blockIdx.x * blockDim.x + threadIdx.x;
  if (i < rows * cols) {
    int rr = i / cols, cc = i - rr * cols;
    o[(size_t)cc * rows + rr] = (_Float16)w[i];
  }
}
// conv_w OIH [64][32][2]  ->  f16 [K=64][N=64], K index = kw*32 + i
__global__ void k_cvt_conv(const float* __restrict__ w, _Float16* __restrict__ o) {
  int i = blockIdx.x * blockDim.x + threadIdx.x;  // i = kk*64 + oc
  if (i < 64 * 64) {
    int kk = i >> 6, oc = i & 63;
    int ci = kk & 31, kw = kk >> 5;
    o[i] = (_Float16)w[oc * 64 + ci * 2 + kw];
  }
}
// f16 [K][N] -> packed B fragments (register order, see b_frag_packed)
__global__ void k_pack_b(const _Float16* __restrict__ B, _Float16* __restrict__ Bp,
                         int K, int N) {
  int i = blockIdx.x * blockDim.x + threadIdx.x;
  if (i < K * N) {
    const int ntiles = N >> 4;
    const int tile = i >> 9;                     // 512 elems per 32x16 tile
    const int lane = (i >> 4) & 31;
    const int j    = i & 15;
    const int kt = tile / ntiles, nt = tile - kt * ntiles;
    const int n = (nt << 4) + (lane & 15);
    const int k = (kt << 5) + (((lane >> 4) & 1) << 3) + ((j >> 3) << 4) + (j & 7);
    Bp[i] = B[(size_t)k * N + n];
  }
}

// ---------------- GCN: xw = (x @ gcn_w) * dinv[node], stored twice ----------
__global__ void __launch_bounds__(256)
k_gcn_gemm(const float* __restrict__ x, const _Float16* __restrict__ Wg,
           const float* __restrict__ dinv,
           float* __restrict__ xws, float* __restrict__ acc) {
  const int lane = threadIdx.x & 31;
  const int wid  = blockIdx.x * 8 + (threadIdx.x >> 5);  // 40000 tiles
  const int row0 = (wid >> 1) * 16;
  const int n0t  = wid & 1;
  const float* rowp = x + (size_t)(row0 + (lane & 15)) * NUM_FEAT;
  v8f d = {};
#pragma unroll
  for (int kt = 0; kt < 4; ++kt) {
    v16h a = a_frag_f32(rowp, kt * 32, lane);
    v16h b = b_frag_packed(Wg, 2, kt, n0t, lane);
    d = wmma_f16(a, b, d);
  }
  const int col = n0t * 16 + (lane & 15);
  const int rb  = (lane >> 4) * 8;
#pragma unroll
  for (int r = 0; r < 8; ++r) {
    const int row = row0 + rb + r;
    const float v = d[r] * dinv[row >> 5];       // node = row / SEQ_LEN
    xws[(size_t)row * GCN_H + col] = v;          // scatter source
    acc[(size_t)row * GCN_H + col] = v;          // self-loop term (init)
  }
}

// ---------------- edge scatter: acc[dst] += xws[src]  (1024 f32 per edge) ---
__global__ void k_scatter(const int* __restrict__ src, const int* __restrict__ dst,
                          const float* __restrict__ xws, float* __restrict__ acc) {
  const int e = blockIdx.x;
  const int s = src[e], dn = dst[e];
  const float* sp = xws + (size_t)s * (SEQ_LEN * GCN_H);
  float*       dp = acc + (size_t)dn * (SEQ_LEN * GCN_H);
  __builtin_prefetch(sp + threadIdx.x, 0, 0);    // global_prefetch_b8
  for (int i = threadIdx.x; i < SEQ_LEN * GCN_H; i += blockDim.x)
    atomicAdd(dp + i, sp[i]);
}

// ---------------- h16 = relu(acc * dinv[dst] + gcn_b) as f16 ----------------
__global__ void k_gcn_act(const float* __restrict__ acc, const float* __restrict__ dinv,
                          const float* __restrict__ gb, _Float16* __restrict__ h16) {
  int i = blockIdx.x * blockDim.x + threadIdx.x;
  if (i < M_GCN * GCN_H) {
    const int c = i & (GCN_H - 1);
    const int n = i >> 10;                       // i / (SEQ_LEN*GCN_H)
    const float v = acc[i] * dinv[n] + gb[c];
    h16[i] = (_Float16)fmaxf(v, 0.0f);
  }
}

// ---------------- Conv1d(32->64,k=2) as GEMM [310000x64]@[64x64] + relu -----
__global__ void __launch_bounds__(256)
k_conv_gemm(const _Float16* __restrict__ h16, const _Float16* __restrict__ Wc,
            const float* __restrict__ cb, _Float16* __restrict__ seq16) {
  const int lane = threadIdx.x & 31;
  const int wid  = blockIdx.x * 8 + (threadIdx.x >> 5);
  if (wid >= (M_CONV / 16) * 4) return;          // uniform per wave
  const int row0 = (wid >> 2) * 16;
  const int nt   = wid & 3;
  const int g    = row0 + (lane & 15);
  const int node = g / TS, tt = g - node * TS;
  // rows t and t+1 are contiguous in h16 => K=64 contiguous window
  const _Float16* rowp = h16 + ((size_t)node * SEQ_LEN + tt) * GCN_H;
  v8f d = {};
#pragma unroll
  for (int kt = 0; kt < 2; ++kt) {
    v16h a = a_frag_f16(rowp, kt * 32, lane);
    v16h b = b_frag_packed(Wc, 4, kt, nt, lane);
    d = wmma_f16(a, b, d);
  }
  const int col = nt * 16 + (lane & 15);
  const int rb  = (lane >> 4) * 8;
#pragma unroll
  for (int r = 0; r < 8; ++r) {
    const float v = fmaxf(d[r] + cb[col], 0.0f);
    seq16[(size_t)(row0 + rb + r) * CONV_H + col] = (_Float16)v;
  }
}

// ---------------- zero-init LSTM state --------------------------------------
__global__ void k_zero_state(_Float16* __restrict__ h, float* __restrict__ c) {
  int i = blockIdx.x * blockDim.x + threadIdx.x;
  if (i < N_NODES * LSTM_H) { h[i] = (_Float16)0.0f; c[i] = 0.0f; }
}

// ---------------- one LSTM step: fused input+recurrent GEMM (K=192) ---------
__global__ void __launch_bounds__(256)
k_lstm_step(const _Float16* __restrict__ seq16, int t,
            const _Float16* __restrict__ Wih,   // packed, ntiles_n = 32
            const _Float16* __restrict__ Whh,   // packed, ntiles_n = 32
            const float* __restrict__ b_ih, const float* __restrict__ b_hh,
            const _Float16* __restrict__ hprev, _Float16* __restrict__ hnext,
            float* __restrict__ cstate) {
  __shared__ float gates[16][4 * LSTM_H];        // 32 KB
  const int lane = threadIdx.x & 31;
  const int wave = threadIdx.x >> 5;
  const int row0 = blockIdx.x * 16;
  const int m    = lane & 15;
  const _Float16* seqrow = seq16 + ((size_t)(row0 + m) * TS + t) * CONV_H;
  const _Float16* hrow   = hprev + (size_t)(row0 + m) * LSTM_H;
  // A fragments are shared by all 4 N-tiles of this wave: load once (48 VGPRs)
  v16h as0 = a_frag_f16(seqrow, 0,  lane);
  v16h as1 = a_frag_f16(seqrow, 32, lane);
  v16h ah0 = a_frag_f16(hrow,   0,  lane);
  v16h ah1 = a_frag_f16(hrow,  32,  lane);
  v16h ah2 = a_frag_f16(hrow,  64,  lane);
  v16h ah3 = a_frag_f16(hrow,  96,  lane);
#pragma unroll
  for (int q = 0; q < 4; ++q) {                  // 8 waves x 4 N-tiles = 512 cols
    const int nt = wave * 4 + q;
    v8f d = {};
    d = wmma_f16(as0, b_frag_packed(Wih, 32, 0, nt, lane), d);
    d = wmma_f16(as1, b_frag_packed(Wih, 32, 1, nt, lane), d);
    d = wmma_f16(ah0, b_frag_packed(Whh, 32, 0, nt, lane), d);
    d = wmma_f16(ah1, b_frag_packed(Whh, 32, 1, nt, lane), d);
    d = wmma_f16(ah2, b_frag_packed(Whh, 32, 2, nt, lane), d);
    d = wmma_f16(ah3, b_frag_packed(Whh, 32, 3, nt, lane), d);
    const int col = nt * 16 + m;
    const int rb  = (lane >> 4) * 8;
#pragma unroll
    for (int r = 0; r < 8; ++r) gates[rb + r][col] = d[r];
  }
  __syncthreads();
  for (int idx = threadIdx.x; idx < 16 * LSTM_H; idx += 256) {
    const int r = idx >> 7, j = idx & (LSTM_H - 1);
    const size_t o = (size_t)(row0 + r) * LSTM_H + j;
    const float gi = gates[r][j]              + b_ih[j]              + b_hh[j];
    const float gf = gates[r][LSTM_H + j]     + b_ih[LSTM_H + j]     + b_hh[LSTM_H + j];
    const float gg = gates[r][2 * LSTM_H + j] + b_ih[2 * LSTM_H + j] + b_hh[2 * LSTM_H + j];
    const float go = gates[r][3 * LSTM_H + j] + b_ih[3 * LSTM_H + j] + b_hh[3 * LSTM_H + j];
    const float si = 1.0f / (1.0f + __expf(-gi));
    const float sf = 1.0f / (1.0f + __expf(-gf));
    const float so = 1.0f / (1.0f + __expf(-go));
    const float cn = sf * cstate[o] + si * tanhf(gg);
    cstate[o] = cn;
    hnext[o]  = (_Float16)(so * tanhf(cn));
  }
}

// ---------------- MLP head: relu(h@fc1+b1)@fc2+b2, fused through LDS --------
__global__ void __launch_bounds__(256)
k_head(const _Float16* __restrict__ hlast,
       const _Float16* __restrict__ Wf1, const float* __restrict__ b1,
       const _Float16* __restrict__ Wf2, const float* __restrict__ b2,
       float* __restrict__ out) {
  __shared__ _Float16 t1[16][64];
  const int lane = threadIdx.x & 31;
  const int wave = threadIdx.x >> 5;
  const int row0 = blockIdx.x * 16;
  const int m    = lane & 15;
  if (wave < 4) {                                // GEMM1: [16x128]@[128x64]
    const int nt = wave;
    const _Float16* rowp = hlast + (size_t)(row0 + m) * LSTM_H;
    v8f d = {};
#pragma unroll
    for (int kt = 0; kt < 4; ++kt) {
      v16h a = a_frag_f16(rowp, kt * 32, lane);
      v16h b = b_frag_packed(Wf1, 4, kt, nt, lane);
      d = wmma_f16(a, b, d);
    }
    const int col = nt * 16 + m;
    const int rb  = (lane >> 4) * 8;
#pragma unroll
    for (int r = 0; r < 8; ++r)
      t1[rb + r][col] = (_Float16)fmaxf(d[r] + b1[col], 0.0f);
  }
  __syncthreads();
  {                                              // GEMM2: [16x64]@[64x128]
    const int nt = wave;
    const _Float16* rowp = &t1[m][0];
    v8f d = {};
#pragma unroll
    for (int kt = 0; kt < 2; ++kt) {
      v16h a = a_frag_f16(rowp, kt * 32, lane);
      v16h b = b_frag_packed(Wf2, 8, kt, nt, lane);
      d = wmma_f16(a, b, d);
    }
    const int col = nt * 16 + m;
    const int rb  = (lane >> 4) * 8;
#pragma unroll
    for (int r = 0; r < 8; ++r)
      out[(size_t)(row0 + rb + r) * NUM_FEAT + col] = d[r] + b2[col];
  }
}

// ---------------- host-side orchestration -----------------------------------
extern "C" void kernel_launch(void* const* d_in, const int* in_sizes, int n_in,
                              void* d_out, int out_size, void* d_ws, size_t ws_size,
                              hipStream_t stream) {
  const float* x      = (const float*)d_in[0];
  const int*   eidx   = (const int*)d_in[1];
  const int    E      = in_sizes[1] / 2;
  const int*   esrc   = eidx;
  const int*   edst   = eidx + E;
  const float* gcn_w  = (const float*)d_in[2];
  const float* gcn_b  = (const float*)d_in[3];
  const float* conv_w = (const float*)d_in[4];
  const float* conv_b = (const float*)d_in[5];
  const float* w_ih   = (const float*)d_in[6];
  const float* w_hh   = (const float*)d_in[7];
  const float* b_ih   = (const float*)d_in[8];
  const float* b_hh   = (const float*)d_in[9];
  const float* fc1_w  = (const float*)d_in[10];
  const float* fc1_b  = (const float*)d_in[11];
  const float* fc2_w  = (const float*)d_in[12];
  const float* fc2_b  = (const float*)d_in[13];
  float* out = (float*)d_out;

  // -- bump allocator over d_ws (with region aliasing) --
  char* base = (char*)d_ws;
  size_t off = 0;
  auto take = [&](size_t bytes) -> void* {
    void* r = base + off;
    off += (bytes + 255) & ~(size_t)255;
    return r;
  };
  int*       deg    = (int*)take(N_NODES * 4);
  float*     dinv   = (float*)take(N_NODES * 4);
  _Float16*  stg    = (_Float16*)take(128 * 512 * 2);   // unpacked staging (reused)
  _Float16*  Wg16   = (_Float16*)take(NUM_FEAT * GCN_H * 2);
  _Float16*  Wc16   = (_Float16*)take(64 * 64 * 2);
  _Float16*  Wih16  = (_Float16*)take(64 * 512 * 2);
  _Float16*  Whh16  = (_Float16*)take(128 * 512 * 2);
  _Float16*  Wf116  = (_Float16*)take(128 * 64 * 2);
  _Float16*  Wf216  = (_Float16*)take(64 * 128 * 2);
  char* bufA = (char*)take((size_t)M_GCN * GCN_H * 4);  // xws, later seq16
  char* bufB = (char*)take((size_t)M_GCN * GCN_H * 4);  // acc, later LSTM state
  _Float16* h16 = (_Float16*)take((size_t)M_GCN * GCN_H * 2);

  float*    xws   = (float*)bufA;
  float*    acc   = (float*)bufB;
  _Float16* seq16 = (_Float16*)bufA;                    // alias: xws dead after scatter
  _Float16* hA    = (_Float16*)bufB;                    // alias: acc dead after act
  _Float16* hB    = (_Float16*)(bufB + 4 * 1024 * 1024);
  float*    cbuf  = (float*)(bufB + 8 * 1024 * 1024);

  // 1) degrees + dinv
  k_init_deg<<<(N_NODES + 255) / 256, 256, 0, stream>>>(deg);
  k_deg_count<<<(E + 255) / 256, 256, 0, stream>>>(edst, E, deg);
  k_dinv<<<(N_NODES + 255) / 256, 256, 0, stream>>>(deg, dinv);

  // 2) weight conversion + fragment packing (stream-ordered, staging reused)
  auto pack = [&](const _Float16* src16, _Float16* dst, int K, int N) {
    k_pack_b<<<(K * N + 255) / 256, 256, 0, stream>>>(src16, dst, K, N);
  };
  k_cvt<<<(NUM_FEAT * GCN_H + 255) / 256, 256, 0, stream>>>(gcn_w, stg, NUM_FEAT * GCN_H);
  pack(stg, Wg16, NUM_FEAT, GCN_H);
  k_cvt_conv<<<(64 * 64 + 255) / 256, 256, 0, stream>>>(conv_w, stg);
  pack(stg, Wc16, 64, 64);
  k_cvt_T<<<(512 * 64 + 255) / 256, 256, 0, stream>>>(w_ih, stg, 512, 64);
  pack(stg, Wih16, 64, 512);
  k_cvt_T<<<(512 * 128 + 255) / 256, 256, 0, stream>>>(w_hh, stg, 512, 128);
  pack(stg, Whh16, 128, 512);
  k_cvt<<<(128 * 64 + 255) / 256, 256, 0, stream>>>(fc1_w, stg, 128 * 64);
  pack(stg, Wf116, 128, 64);
  k_cvt<<<(64 * 128 + 255) / 256, 256, 0, stream>>>(fc2_w, stg, 64 * 128);
  pack(stg, Wf216, 64, 128);

  // 3) GCN transform GEMM (pre-scaled by dinv[src]; self-loop seeds acc)
  k_gcn_gemm<<<(M_GCN / 16) * 2 / 8, 256, 0, stream>>>(x, Wg16, dinv, xws, acc);

  // 4) edge scatter (one block per edge, f32 atomics)
  k_scatter<<<E, 256, 0, stream>>>(esrc, edst, xws, acc);

  // 5) relu(acc * dinv[dst] + b) -> f16
  k_gcn_act<<<(M_GCN * GCN_H + 255) / 256, 256, 0, stream>>>(acc, dinv, gcn_b, h16);

  // 6) Conv1d as GEMM + relu -> seq16 (aliases bufA)
  {
    const int tiles = (M_CONV / 16) * 4;
    k_conv_gemm<<<(tiles + 7) / 8, 256, 0, stream>>>(h16, Wc16, conv_b, seq16);
  }

  // 7) LSTM: zero state then 31 fused steps (ping-pong h)
  k_zero_state<<<(N_NODES * LSTM_H + 255) / 256, 256, 0, stream>>>(hA, cbuf);
  for (int t = 0; t < TS; ++t) {
    const _Float16* hin  = (t & 1) ? hB : hA;
    _Float16*       hout = (t & 1) ? hA : hB;
    k_lstm_step<<<N_NODES / 16, 256, 0, stream>>>(seq16, t, Wih16, Whh16,
                                                  b_ih, b_hh, hin, hout, cbuf);
  }
  const _Float16* hlast = hB;  // TS=31 steps, last (t=30, even) writes hB

  // 8) MLP head
  k_head<<<N_NODES / 16, 256, 0, stream>>>(hlast, Wf116, fc1_b, Wf216, fc2_b, out);
}
